// MultiheadAttention_66958540144669
// MI455X (gfx1250) — matrix operand order repaired and checked by reading, not compiled, not run on hardware
//
#include <hip/hip_runtime.h>

#define D_MODEL 1024
#define N_HEADS 16
#define HEAD_DIM 64
#define BATCH 4
#define SEQ 2048
#define ROWS (BATCH * SEQ)            // 8192
#define ATT_SCALE 0.125f              // 64^-0.5

typedef __bf16 bf16;
typedef __attribute__((ext_vector_type(16))) __bf16 v16bf;
typedef __attribute__((ext_vector_type(8)))  float  v8f;
typedef __attribute__((ext_vector_type(4)))  float  f32x4;
typedef __attribute__((ext_vector_type(4)))  unsigned int u32x4;
typedef __attribute__((ext_vector_type(2)))  unsigned int u32x2;

union PackA { bf16 h[16]; u32x4 q[2]; };
union Pack8 { bf16 h[8];  u32x4 q; };
union Pack4 { bf16 h[4];  u32x2 q; };
union FragU { u32x4 q[2]; v16bf v; };

// ---------------- WMMA fragment helpers (wave32 layouts per CDNA5 ISA 7.12.2) ----

// A 16x32 bf16: lane holds row M=lane&15; halves 0-7 = K[kb..kb+7], halves 8-15 =
// K[16+kb..16+kb+7], kb = (lane>>4)*8. Two 16B LDS loads.
__device__ __forceinline__ v16bf ld_a_frag(const bf16* p, int stride, int lane) {
  const int row = lane & 15;
  const int kb  = (lane >> 4) << 3;
  FragU f;
  f.q[0] = *(const u32x4*)(p + row * stride + kb);
  f.q[1] = *(const u32x4*)(p + row * stride + kb + 16);
  return f.v;
}

// B 32x16 bf16 stored in LDS as [N][K]: lane holds col N=lane&15, halves 0-7 =
// K[kb..kb+7], halves 8-15 = K[16+kb..16+kb+7], kb = (lane>>4)*8. Two 16B LDS loads.
__device__ __forceinline__ v16bf ld_b_frag(const bf16* p, int stride, int lane) {
  const int n  = lane & 15;
  const int kb = (lane >> 4) << 3;
  FragU f;
  f.q[0] = *(const u32x4*)(p + n * stride + kb);
  f.q[1] = *(const u32x4*)(p + n * stride + kb + 16);
  return f.v;
}

__device__ __forceinline__ v8f wmma_bf16(v16bf a, v16bf b, v8f c) {
  return __builtin_amdgcn_wmma_f32_16x16x32_bf16(false, a, false, b, (short)0, c,
                                                 false, false);
}

// ---------------- CDNA5 async global->LDS ---------------------------------------

__device__ __forceinline__ void async_b128(unsigned lds_off, const void* base,
                                           unsigned byte_off) {
  unsigned long long b = (unsigned long long)base;
  asm volatile("global_load_async_to_lds_b128 %0, %1, %2"
               :: "v"(lds_off), "v"(byte_off), "s"(b) : "memory");
}
__device__ __forceinline__ void wait_async0() {
  asm volatile("s_wait_asynccnt 0" ::: "memory");
}

__device__ __forceinline__ v8f v8f_zero() {
  v8f z = {0.f, 0.f, 0.f, 0.f, 0.f, 0.f, 0.f, 0.f};
  return z;
}

// ---------------- Kernel 0: fp32 weights -> bf16 workspace ----------------------

__global__ void __launch_bounds__(256) cvtw_kernel(const float* __restrict__ w0,
                                                   const float* __restrict__ w1,
                                                   const float* __restrict__ w2,
                                                   const float* __restrict__ w3,
                                                   bf16* __restrict__ out) {
  const float* srcs[4] = {w0, w1, w2, w3};
  const float* s = srcs[blockIdx.y];
  bf16* d = out + (size_t)blockIdx.y * (D_MODEL * D_MODEL);
  int i = blockIdx.x * 256 + threadIdx.x;   // f32x4 index, 1M/4 = 262144 total
  f32x4 v = ((const f32x4*)s)[i];
  Pack4 p;
  p.h[0] = (bf16)v.x; p.h[1] = (bf16)v.y; p.h[2] = (bf16)v.z; p.h[3] = (bf16)v.w;
  ((u32x2*)d)[i] = p.q;
}

// ---------------- Kernel 1: projection  P = X @ W^T + b  (bf16 out) -------------

__global__ void __launch_bounds__(256) proj_kernel(const float* __restrict__ X,
                                                   const bf16* __restrict__ W,
                                                   const float* __restrict__ bias,
                                                   bf16* __restrict__ Out) {
  extern __shared__ __align__(16) char smem[];
  bf16* Alds = (bf16*)smem;                        // [128][32]
  bf16* Blds = (bf16*)(smem + 128 * 32 * 2);       // [64][32]  ([N][K])
  const unsigned BOFF = 128 * 32 * 2;              // 8192

  const int t = threadIdx.x, lane = t & 31, w = t >> 5;
  const int m0 = blockIdx.x * 128;
  const int n0 = blockIdx.y * 64;

  v8f acc[4] = {v8f_zero(), v8f_zero(), v8f_zero(), v8f_zero()};

  for (int k0 = 0; k0 < D_MODEL; k0 += 32) {
    // stage A: 128x32 fp32 -> bf16  (2 threads per row)
    {
      int row = t >> 1, half = (t & 1) * 16;
      const f32x4* src = (const f32x4*)(X + (size_t)(m0 + row) * D_MODEL + k0 + half);
      f32x4 f0 = src[0], f1 = src[1], f2 = src[2], f3 = src[3];
      PackA p;
      p.h[0] = (bf16)f0.x; p.h[1]  = (bf16)f0.y; p.h[2]  = (bf16)f0.z; p.h[3]  = (bf16)f0.w;
      p.h[4] = (bf16)f1.x; p.h[5]  = (bf16)f1.y; p.h[6]  = (bf16)f1.z; p.h[7]  = (bf16)f1.w;
      p.h[8] = (bf16)f2.x; p.h[9]  = (bf16)f2.y; p.h[10] = (bf16)f2.z; p.h[11] = (bf16)f2.w;
      p.h[12]= (bf16)f3.x; p.h[13] = (bf16)f3.y; p.h[14] = (bf16)f3.z; p.h[15] = (bf16)f3.w;
      u32x4* dst = (u32x4*)(Alds + row * 32 + half);
      dst[0] = p.q[0]; dst[1] = p.q[1];
    }
    // stage B (bf16 weights): async, 64 rows x 64B, 16B per lane
    {
      int row = t >> 2, seg = t & 3;
      unsigned goff = ((unsigned)(n0 + row) * D_MODEL + (unsigned)k0) * 2 + seg * 16;
      async_b128(BOFF + row * 64 + seg * 16, W, goff);
    }
    wait_async0();
    __syncthreads();

    v16bf a = ld_a_frag(Alds + w * 16 * 32, 32, lane);
#pragma unroll
    for (int nb = 0; nb < 4; ++nb) {
      v16bf b = ld_b_frag(Blds + nb * 16 * 32, 32, lane);
      acc[nb] = wmma_bf16(a, b, acc[nb]);
    }
    __syncthreads();
  }

  const int mrow = m0 + w * 16 + ((lane >> 4) << 3);
#pragma unroll
  for (int nb = 0; nb < 4; ++nb) {
    int n = n0 + nb * 16 + (lane & 15);
    float bv = bias[n];
#pragma unroll
    for (int r = 0; r < 8; ++r)
      Out[(size_t)(mrow + r) * D_MODEL + n] = (bf16)(acc[nb][r] + bv);
  }
}

// ---------------- Kernel 2: flash attention -> ctx (bf16) + per-row (m,l) -------
// grid (B*H, S/128). Each wave owns 16 q rows; streams 64-key tiles of K/V.

__global__ void __launch_bounds__(256) flash_kernel(const bf16* __restrict__ Qb,
                                                    const bf16* __restrict__ Kb,
                                                    const bf16* __restrict__ Vb,
                                                    bf16* __restrict__ Ctx,
                                                    float* __restrict__ ML) {
  extern __shared__ __align__(16) char smem[];
  bf16* Qlds = (bf16*)smem;                    // [128][64]
  bf16* Klds = (bf16*)(smem + 16384);          // [64][64]  rows = key, cols = dh
  bf16* Vlds = (bf16*)(smem + 24576);          // [64][64]  transposed: [dh][key]
  bf16* Plds = (bf16*)(smem + 32768);          // 8 waves x [16][64]
  const unsigned KOFF = 16384;

  const int t = threadIdx.x, lane = t & 31, w = t >> 5;
  const int bh = blockIdx.x, b = bh >> 4, h = bh & 15;
  const int q0 = blockIdx.y * 128;
  const int hcol = h * HEAD_DIM;
  bf16* Pw = Plds + w * 16 * 64;               // wave-private P tile

  // stage Q once: 128 rows x 128B (async)
#pragma unroll
  for (int i = 0; i < 4; ++i) {
    int idx = t + i * 256;
    int row = idx >> 3, seg = idx & 7;
    unsigned goff = ((unsigned)(b * SEQ + q0 + row) * D_MODEL + hcol) * 2 + seg * 16;
    async_b128((unsigned)(row * 128 + seg * 16), Qb, goff);
  }

  v8f acc[4] = {v8f_zero(), v8f_zero(), v8f_zero(), v8f_zero()};
  float m[8], l[8];
#pragma unroll
  for (int r = 0; r < 8; ++r) { m[r] = -1e30f; l[r] = 0.f; }

  for (int k0 = 0; k0 < SEQ; k0 += 64) {
    // stage K: 64 rows x 128B (async)
#pragma unroll
    for (int i = 0; i < 2; ++i) {
      int idx = t + i * 256;
      int row = idx >> 3, seg = idx & 7;
      unsigned goff = ((unsigned)(b * SEQ + k0 + row) * D_MODEL + hcol) * 2 + seg * 16;
      async_b128(KOFF + (unsigned)(row * 128 + seg * 16), Kb, goff);
    }
    // stage V transposed into [dh][key]
#pragma unroll
    for (int i = 0; i < 2; ++i) {
      int idx = t + i * 256;                   // 512 = 64 keys x 8 dh-groups
      int key = idx >> 3, dh0 = (idx & 7) * 8;
      Pack8 pv;
      pv.q = *(const u32x4*)(Vb + (size_t)(b * SEQ + k0 + key) * D_MODEL + hcol + dh0);
#pragma unroll
      for (int j = 0; j < 8; ++j) Vlds[(dh0 + j) * 64 + key] = pv.h[j];
    }
    wait_async0();
    __syncthreads();

    // scores tile s[16 x 64] = q @ k^T
    v8f s[4] = {v8f_zero(), v8f_zero(), v8f_zero(), v8f_zero()};
#pragma unroll
    for (int kk = 0; kk < HEAD_DIM; kk += 32) {
      v16bf a = ld_a_frag(Qlds + w * 16 * 64 + kk, 64, lane);
#pragma unroll
      for (int nb = 0; nb < 4; ++nb) {
        v16bf bf = ld_b_frag(Klds + nb * 16 * 64 + kk, 64, lane);
        s[nb] = wmma_bf16(a, bf, s[nb]);
      }
    }

    // online softmax update; row M = (lane>>4)*8 + r lives in one 16-lane group
#pragma unroll
    for (int r = 0; r < 8; ++r) {
      float sv0 = s[0][r] * ATT_SCALE, sv1 = s[1][r] * ATT_SCALE;
      float sv2 = s[2][r] * ATT_SCALE, sv3 = s[3][r] * ATT_SCALE;
      float rmax = fmaxf(fmaxf(sv0, sv1), fmaxf(sv2, sv3));
      rmax = fmaxf(rmax, __shfl_xor(rmax, 1, 16));
      rmax = fmaxf(rmax, __shfl_xor(rmax, 2, 16));
      rmax = fmaxf(rmax, __shfl_xor(rmax, 4, 16));
      rmax = fmaxf(rmax, __shfl_xor(rmax, 8, 16));
      float mnew = fmaxf(m[r], rmax);
      float corr = __expf(m[r] - mnew);
      float e0 = __expf(sv0 - mnew), e1 = __expf(sv1 - mnew);
      float e2 = __expf(sv2 - mnew), e3 = __expf(sv3 - mnew);
      float rsum = e0 + e1 + e2 + e3;
      rsum += __shfl_xor(rsum, 1, 16);
      rsum += __shfl_xor(rsum, 2, 16);
      rsum += __shfl_xor(rsum, 4, 16);
      rsum += __shfl_xor(rsum, 8, 16);
      l[r] = l[r] * corr + rsum;
      m[r] = mnew;
      acc[0][r] *= corr; acc[1][r] *= corr;
      acc[2][r] *= corr; acc[3][r] *= corr;
      // write P (exp probs) to wave-private LDS in row-major [16][64]
      int prow = ((lane >> 4) << 3) + r;
      int pcol = lane & 15;
      Pw[prow * 64 + pcol +  0] = (bf16)e0;
      Pw[prow * 64 + pcol + 16] = (bf16)e1;
      Pw[prow * 64 + pcol + 32] = (bf16)e2;
      Pw[prow * 64 + pcol + 48] = (bf16)e3;
    }

    // ctx += P @ V  (same-wave LDS ops are in-order, no barrier needed for Pw)
#pragma unroll
    for (int kk = 0; kk < 64; kk += 32) {
      v16bf a = ld_a_frag(Pw + kk, 64, lane);
#pragma unroll
      for (int nb = 0; nb < 4; ++nb) {
        v16bf bf = ld_b_frag(Vlds + nb * 16 * 64 + kk, 64, lane);
        acc[nb] = wmma_bf16(a, bf, acc[nb]);
      }
    }
    __syncthreads();   // K/V tiles consumed before next stage overwrites
  }

  // normalize and write ctx (merged-head layout), plus per-row (m, l) stats
#pragma unroll
  for (int r = 0; r < 8; ++r) {
    float inv = 1.0f / l[r];
    acc[0][r] *= inv; acc[1][r] *= inv; acc[2][r] *= inv; acc[3][r] *= inv;
  }
  const int qrow = q0 + w * 16 + ((lane >> 4) << 3);
#pragma unroll
  for (int nb = 0; nb < 4; ++nb) {
    int dh = nb * 16 + (lane & 15);
#pragma unroll
    for (int r = 0; r < 8; ++r)
      Ctx[(size_t)(b * SEQ + qrow + r) * D_MODEL + hcol + dh] = (bf16)acc[nb][r];
  }
  if ((lane & 15) == 0) {
#pragma unroll
    for (int r = 0; r < 8; ++r) {
      size_t row = (size_t)bh * SEQ + qrow + r;
      ML[row * 2 + 0] = m[r];
      ML[row * 2 + 1] = l[r];
    }
  }
}

// ---------------- Kernel 3: weights = exp(q@k^T*scale - m)/l  -> d_out ----------
// Recomputes scores (compute is free; we are HBM-bound) so the 1.07GB weights
// array is written exactly once and never re-read.

__global__ void __launch_bounds__(256) weights_kernel(const bf16* __restrict__ Qb,
                                                      const bf16* __restrict__ Kb,
                                                      const float* __restrict__ ML,
                                                      float* __restrict__ Wgt) {
  extern __shared__ __align__(16) char smem[];
  bf16* Qlds = (bf16*)smem;                       // [128][64]
  bf16* Klds = (bf16*)(smem + 128 * 64 * 2);      // [64][64]
  const unsigned KOFF = 128 * 64 * 2;             // 16384

  const int t = threadIdx.x, lane = t & 31, w = t >> 5;
  const int bh = blockIdx.x, b = bh >> 4, h = bh & 15;
  const int q0 = blockIdx.y * 128;
  const int k0 = blockIdx.z * 64;
  const int hcol = h * HEAD_DIM;

#pragma unroll
  for (int i = 0; i < 4; ++i) {
    int idx = t + i * 256;
    int row = idx >> 3, seg = idx & 7;
    unsigned goff = ((unsigned)(b * SEQ + q0 + row) * D_MODEL + hcol) * 2 + seg * 16;
    async_b128((unsigned)(row * 128 + seg * 16), Qb, goff);
  }
#pragma unroll
  for (int i = 0; i < 2; ++i) {
    int idx = t + i * 256;
    int row = idx >> 3, seg = idx & 7;
    unsigned goff = ((unsigned)(b * SEQ + k0 + row) * D_MODEL + hcol) * 2 + seg * 16;
    async_b128(KOFF + (unsigned)(row * 128 + seg * 16), Kb, goff);
  }
  wait_async0();
  __syncthreads();

  v8f acc[4] = {v8f_zero(), v8f_zero(), v8f_zero(), v8f_zero()};
#pragma unroll
  for (int kk = 0; kk < HEAD_DIM; kk += 32) {
    v16bf a = ld_a_frag(Qlds + w * 16 * 64 + kk, 64, lane);
#pragma unroll
    for (int nb = 0; nb < 4; ++nb) {
      v16bf bfr = ld_b_frag(Klds + nb * 16 * 64 + kk, 64, lane);
      acc[nb] = wmma_bf16(a, bfr, acc[nb]);
    }
  }

  const int qrow = q0 + w * 16 + ((lane >> 4) << 3);
#pragma unroll
  for (int r = 0; r < 8; ++r) {
    size_t row = (size_t)bh * SEQ + qrow + r;
    float mr   = ML[row * 2 + 0];
    float invl = 1.0f / ML[row * 2 + 1];
#pragma unroll
    for (int nb = 0; nb < 4; ++nb) {
      int col = k0 + nb * 16 + (lane & 15);
      Wgt[row * SEQ + col] = __expf(acc[nb][r] * ATT_SCALE - mr) * invl;
    }
  }
}

// ---------------- Kernel 4: out = ctx @ W_o^T + b_o  (fp32 -> d_out) ------------

__global__ void __launch_bounds__(256) outproj_kernel(const bf16* __restrict__ Xb,
                                                      const bf16* __restrict__ W,
                                                      const float* __restrict__ bias,
                                                      float* __restrict__ Out) {
  extern __shared__ __align__(16) char smem[];
  bf16* Alds = (bf16*)smem;                  // [128][32]
  const unsigned BOFF = 8192;
  bf16* Blds = (bf16*)(smem + BOFF);         // [64][32]

  const int t = threadIdx.x, lane = t & 31, w = t >> 5;
  const int m0 = blockIdx.x * 128;
  const int n0 = blockIdx.y * 64;

  v8f acc[4] = {v8f_zero(), v8f_zero(), v8f_zero(), v8f_zero()};

  for (int k0 = 0; k0 < D_MODEL; k0 += 32) {
#pragma unroll
    for (int i = 0; i < 2; ++i) {
      int idx = t + i * 256;
      int row = idx >> 2, seg = idx & 3;
      unsigned goff = ((unsigned)(m0 + row) * D_MODEL + (unsigned)k0) * 2 + seg * 16;
      async_b128((unsigned)(row * 64 + seg * 16), Xb, goff);
    }
    {
      int row = t >> 2, seg = t & 3;
      unsigned goff = ((unsigned)(n0 + row) * D_MODEL + (unsigned)k0) * 2 + seg * 16;
      async_b128(BOFF + (unsigned)(row * 64 + seg * 16), W, goff);
    }
    wait_async0();
    __syncthreads();

    v16bf a = ld_a_frag(Alds + w * 16 * 32, 32, lane);
#pragma unroll
    for (int nb = 0; nb < 4; ++nb) {
      v16bf b = ld_b_frag(Blds + nb * 16 * 32, 32, lane);
      acc[nb] = wmma_bf16(a, b, acc[nb]);
    }
    __syncthreads();
  }

  const int mrow = m0 + w * 16 + ((lane >> 4) << 3);
#pragma unroll
  for (int nb = 0; nb < 4; ++nb) {
    int n = n0 + nb * 16 + (lane & 15);
    float bv = bias[n];
#pragma unroll
    for (int r = 0; r < 8; ++r)
      Out[(size_t)(mrow + r) * D_MODEL + n] = acc[nb][r] + bv;
  }
}

// ---------------- launcher -------------------------------------------------------

extern "C" void kernel_launch(void* const* d_in, const int* in_sizes, int n_in,
                              void* d_out, int out_size, void* d_ws, size_t ws_size,
                              hipStream_t stream) {
  const float* Q   = (const float*)d_in[0];
  const float* K   = (const float*)d_in[1];
  const float* V   = (const float*)d_in[2];
  const float* W_q = (const float*)d_in[3];
  const float* b_q = (const float*)d_in[4];
  const float* W_k = (const float*)d_in[5];
  const float* b_k = (const float*)d_in[6];
  const float* W_v = (const float*)d_in[7];
  const float* b_v = (const float*)d_in[8];
  const float* W_o = (const float*)d_in[9];
  const float* b_o = (const float*)d_in[10];

  // ws (73 MB): qb | kb | vb | ctx (bf16, 16MB each) | 4x bf16 weights (8MB) | ML (1MB)
  char* ws = (char*)d_ws;
  bf16* qb   = (bf16*)(ws);
  bf16* kb   = (bf16*)(ws + ((size_t)16 << 20));
  bf16* vb   = (bf16*)(ws + ((size_t)32 << 20));
  bf16* ctxb = (bf16*)(ws + ((size_t)48 << 20));
  bf16* wb   = (bf16*)(ws + ((size_t)64 << 20));
  float* ml  = (float*)(ws + ((size_t)72 << 20));
  const size_t WSZ = (size_t)D_MODEL * D_MODEL;

  float* out = (float*)d_out;
  float* wgt = out + (size_t)ROWS * D_MODEL;   // weights region follows `out`

  dim3 blk(256);
  cvtw_kernel<<<dim3(1024, 4), blk, 0, stream>>>(W_q, W_k, W_v, W_o, wb);

  proj_kernel<<<dim3(64, 16), blk, 12288, stream>>>(Q, wb + 0 * WSZ, b_q, qb);
  proj_kernel<<<dim3(64, 16), blk, 12288, stream>>>(K, wb + 1 * WSZ, b_k, kb);
  proj_kernel<<<dim3(64, 16), blk, 12288, stream>>>(V, wb + 2 * WSZ, b_v, vb);

  flash_kernel<<<dim3(64, 16), blk, 49152, stream>>>(qb, kb, vb, ctxb, ml);
  weights_kernel<<<dim3(64, 16, 32), blk, 24576, stream>>>(qb, kb, ml, wgt);
  outproj_kernel<<<dim3(64, 16), blk, 12288, stream>>>(ctxb, wb + 3 * WSZ, b_o, out);
}
